// LSTM_Attention_model_80977313399719
// MI455X (gfx1250) — compile-verified, matrix operand order checked by reading, not run on hardware
//
#include <hip/hip_runtime.h>
#include <hip/hip_bf16.h>

// ---------------------------------------------------------------------------
// MI455X (gfx1250) implementation of the 3-GRU + prefix-attention model.
//  - Input projections: f16 WMMA GEMMs (v_wmma_f32_16x16x32_f16), fp32 accum.
//  - Recurrences: 16 sequences per workgroup, full scan in one kernel; Whh
//    kept in VGPRs as pre-built WMMA B-fragments (loaded once, reused per step).
//  - Prefix attention folded into an O(T*H) running-softmax scan per row.
// ---------------------------------------------------------------------------

typedef __attribute__((ext_vector_type(16))) _Float16 v16h;
typedef __attribute__((ext_vector_type(8)))  _Float16 v8h;
typedef __attribute__((ext_vector_type(8)))  float    v8f;

static constexpr int kB    = 256;
static constexpr int kT    = 512;
static constexpr int kM    = kB * kT;     // 131072 rows
static constexpr int kH    = 128;
static constexpr int kG    = 384;         // 3*H gates (r,z,n)

// K (real) and K padded to multiple of 32 per GRU
static constexpr int kKsh = 70,  kKshP = 96;
static constexpr int kKt1 = 172, kKt1P = 192;
static constexpr int kKt2 = 199, kKt2P = 224;

// Workspace layout (bytes, all 256B aligned). Total ~400 MB.
static constexpr size_t OFF_TT    = 0;                         // 131072*10*4
static constexpr size_t OFF_X16   = OFF_TT    + 5242880;       // 131072*224*2
static constexpr size_t OFF_GI    = OFF_X16   + 58720256;      // 131072*384*4
static constexpr size_t OFF_HSH   = OFF_GI    + 201326592;     // 131072*128*4
static constexpr size_t OFF_HT    = OFF_HSH   + 67108864;      // 131072*128*4
static constexpr size_t OFF_WIH16 = OFF_HT    + 67108864;      // 384*224*2
static constexpr size_t OFF_WHH16 = OFF_WIH16 + 172032;        // 384*128*2

__device__ __forceinline__ float sigmoidf_(float x) {
    return 1.0f / (1.0f + __expf(-x));
}

// ---------------------------------------------------------------------------
// RelTemporalEncoding: tt[r,:] = emb[delta[r]] @ rte_W^T + rte_b   (10x10)
// ---------------------------------------------------------------------------
__global__ void rte_kernel(const int* __restrict__ delta,
                           const float* __restrict__ emb,
                           const float* __restrict__ W,
                           const float* __restrict__ bias,
                           float* __restrict__ tt) {
    int r = blockIdx.x * blockDim.x + threadIdx.x;
    if (r >= kM) return;
    const float* er = emb + (size_t)delta[r] * 10;
    float ev[10];
#pragma unroll
    for (int i = 0; i < 10; ++i) ev[i] = er[i];
#pragma unroll
    for (int o = 0; o < 10; ++o) {
        float acc = bias[o];
#pragma unroll
        for (int i = 0; i < 10; ++i) acc += ev[i] * W[o * 10 + i];
        tt[(size_t)r * 10 + o] = acc;
    }
}

// ---------------------------------------------------------------------------
// Convert a [rows x K] f32 weight to [rows x Kp] f16, zero-padding K..Kp.
// ---------------------------------------------------------------------------
__global__ void pack_w_kernel(const float* __restrict__ W, _Float16* __restrict__ W16,
                              int K, int Kp, int rows) {
    int idx = blockIdx.x * blockDim.x + threadIdx.x;
    if (idx >= rows * Kp) return;
    int k = idx % Kp, g = idx / Kp;
    W16[idx] = (_Float16)((k < K) ? W[(size_t)g * K + k] : 0.0f);
}

// ---------------------------------------------------------------------------
// Build padded f16 concat-input matrix X [kM x Kp] for the selected GRU.
// mode 0: [input(27) | nnan(27) | static(6) | tt(10)]                 K=70
// mode 1: [h_sh(128) | nnan(27) | hgb(1) | static(6) | tt(10)]        K=172
// mode 2: [h_sh(128) | nnan(27) | hgb(1) | input(27) | static(6) | tt(10)] K=199
// ---------------------------------------------------------------------------
__global__ void pack_x_kernel(int mode, int Kp,
                              const float* __restrict__ ib,
                              const float* __restrict__ hgb,
                              const float* __restrict__ st,
                              const float* __restrict__ nn,
                              const float* __restrict__ tt,
                              const float* __restrict__ hsh,
                              _Float16* __restrict__ X) {
    size_t idx = (size_t)blockIdx.x * blockDim.x + threadIdx.x;
    size_t total = (size_t)kM * Kp;
    if (idx >= total) return;
    int k = (int)(idx % Kp);
    size_t r = idx / Kp;
    float v = 0.0f;
    if (mode == 0) {
        if      (k < 27) v = ib[r * 27 + k];
        else if (k < 54) v = nn[r * 27 + (k - 27)];
        else if (k < 60) v = st[r * 6 + (k - 54)];
        else if (k < 70) v = tt[r * 10 + (k - 60)];
    } else if (mode == 1) {
        if      (k < 128) v = hsh[r * 128 + k];
        else if (k < 155) v = nn[r * 27 + (k - 128)];
        else if (k < 156) v = hgb[r];
        else if (k < 162) v = st[r * 6 + (k - 156)];
        else if (k < 172) v = tt[r * 10 + (k - 162)];
    } else {
        if      (k < 128) v = hsh[r * 128 + k];
        else if (k < 155) v = nn[r * 27 + (k - 128)];
        else if (k < 156) v = hgb[r];
        else if (k < 183) v = ib[r * 27 + (k - 156)];
        else if (k < 189) v = st[r * 6 + (k - 183)];
        else if (k < 199) v = tt[r * 10 + (k - 189)];
    }
    X[idx] = (_Float16)v;
}

// ---------------------------------------------------------------------------
// gi = X @ W^T + bih.  One wave per 16x16 output tile.
// A-frag layout (ISA 7.12.2): lane L holds row M=L%16; K = (L/16)*8 + {0..7}
// in halfs 0..7 and K+16 in halfs 8..15 of each 32-wide K chunk.
// ---------------------------------------------------------------------------
__global__ void wmma_gemm_kernel(const _Float16* __restrict__ X,   // [kM][Kp]
                                 const _Float16* __restrict__ W16, // [kG][Kp]
                                 const float* __restrict__ bih,    // [kG]
                                 float* __restrict__ gi,           // [kM][kG]
                                 int Kp) {
    const int lane   = threadIdx.x;
    const int lane16 = lane & 15;
    const int halfg  = lane >> 4;
    const int mbase  = blockIdx.x * 16;
    const int n      = blockIdx.y * 16 + lane16;

    v8f c;
    {
        float bv = bih[n];
#pragma unroll
        for (int r = 0; r < 8; ++r) c[r] = bv;
    }
    const int m = mbase + lane16;
    const _Float16* xrow = X + (size_t)m * Kp;
    const _Float16* wrow = W16 + (size_t)n * Kp;

    for (int kc = 0; kc < Kp; kc += 32) {
        int k0 = kc + halfg * 8;
        v8h alo = *(const v8h*)(xrow + k0);
        v8h ahi = *(const v8h*)(xrow + k0 + 16);
        v8h blo = *(const v8h*)(wrow + k0);
        v8h bhi = *(const v8h*)(wrow + k0 + 16);
        v16h a = __builtin_shufflevector(alo, ahi, 0,1,2,3,4,5,6,7,8,9,10,11,12,13,14,15);
        v16h b = __builtin_shufflevector(blo, bhi, 0,1,2,3,4,5,6,7,8,9,10,11,12,13,14,15);
        c = __builtin_amdgcn_wmma_f32_16x16x32_f16(false, a, false, b, (short)0, c,
                                                   false, false);
    }
#pragma unroll
    for (int r = 0; r < 8; ++r) {
        int mm = mbase + r + 8 * halfg;   // C/D layout: VGPR r -> M = r + 8*(lane/16)
        gi[(size_t)mm * kG + n] = c[r];
    }
}

// ---------------------------------------------------------------------------
// GRU scan. Each block owns 16 independent sequences; runs all steps locally.
// gi/h row index = seq*seq_rs + step*step_rs (handles both scan orientations).
// Whh B-fragments are built once into VGPRs (12 x v16h per wave) and reused
// every step; per-step work = 12 WMMAs per wave + fused gate nonlinearity.
// ---------------------------------------------------------------------------
__global__ void __launch_bounds__(256)
gru_scan_kernel(const float* __restrict__ gi,        // [kM][384]
                const _Float16* __restrict__ Whh16,  // [384][128] f16
                const float* __restrict__ bhh,       // [384]
                float* __restrict__ hout,            // [kM][128]
                int nsteps, int seq_rs, int step_rs) {
    const int tid    = threadIdx.x;
    const int wave   = tid >> 5;
    const int lane   = tid & 31;
    const int lane16 = lane & 15;
    const int halfg  = lane >> 4;
    const int seqbase = blockIdx.x * 16;

    __shared__ _Float16 h16[16][128];
    __shared__ float    hprev[16][128];
    __shared__ float    gh[16][384];

    for (int e = tid; e < 16 * 128; e += 256) {
        hprev[e >> 7][e & 127] = 0.0f;
        h16[e >> 7][e & 127]   = (_Float16)0.0f;
    }

    // Preload this wave's B-fragments of Whh (columns [wave*48, wave*48+48)).
    v16h Bf[3][4];
    float bbias[3];
#pragma unroll
    for (int gt = 0; gt < 3; ++gt) {
        int g = wave * 48 + gt * 16 + lane16;
        bbias[gt] = bhh[g];
        const _Float16* wrow = Whh16 + (size_t)g * 128;
#pragma unroll
        for (int kc = 0; kc < 4; ++kc) {
            int k0 = kc * 32 + halfg * 8;
            v8h blo = *(const v8h*)(wrow + k0);
            v8h bhi = *(const v8h*)(wrow + k0 + 16);
            Bf[gt][kc] = __builtin_shufflevector(blo, bhi,
                0,1,2,3,4,5,6,7,8,9,10,11,12,13,14,15);
        }
    }
    __syncthreads();

    for (int s = 0; s < nsteps; ++s) {
        // gh = h_prev @ Whh^T + bhh   (16 x 384)
#pragma unroll
        for (int gt = 0; gt < 3; ++gt) {
            v8f c;
#pragma unroll
            for (int r = 0; r < 8; ++r) c[r] = bbias[gt];
#pragma unroll
            for (int kc = 0; kc < 4; ++kc) {
                int k0 = kc * 32 + halfg * 8;
                v8h alo = *(const v8h*)(&h16[lane16][k0]);
                v8h ahi = *(const v8h*)(&h16[lane16][k0 + 16]);
                v16h a = __builtin_shufflevector(alo, ahi,
                    0,1,2,3,4,5,6,7,8,9,10,11,12,13,14,15);
                c = __builtin_amdgcn_wmma_f32_16x16x32_f16(false, a, false, Bf[gt][kc],
                                                           (short)0, c, false, false);
            }
            int g = wave * 48 + gt * 16 + lane16;
#pragma unroll
            for (int r = 0; r < 8; ++r) gh[r + 8 * halfg][g] = c[r];
        }
        __syncthreads();

        // Prefetch next step's gi rows into cache (global_prefetch_b8).
        if (s + 1 < nsteps && tid < 16) {
            size_t nrow = (size_t)(seqbase + tid) * seq_rs + (size_t)(s + 1) * step_rs;
            __builtin_prefetch(gi + nrow * kG, 0, 1);
        }

        // Fused GRU gate nonlinearity + state update.
        for (int e = tid; e < 16 * 128; e += 256) {
            int mrow = e >> 7, j = e & 127;
            size_t row = (size_t)(seqbase + mrow) * seq_rs + (size_t)s * step_rs;
            const float* gr = gi + row * kG;
            float ir = gr[j], iz = gr[128 + j], in_ = gr[256 + j];
            float hr = gh[mrow][j], hz = gh[mrow][128 + j], hn = gh[mrow][256 + j];
            float rg = sigmoidf_(ir + hr);
            float zg = sigmoidf_(iz + hz);
            float ng = tanhf(in_ + rg * hn);
            float h2 = (1.0f - zg) * ng + zg * hprev[mrow][j];
            hprev[mrow][j] = h2;
            h16[mrow][j]   = (_Float16)h2;
            hout[row * kH + j] = h2;
        }
        __syncthreads();
    }
}

// ---------------------------------------------------------------------------
// Prefix attention as running softmax scan. Scores e_j are t-independent, so
// c_t = cumsum(e^{e_j-M} h_j) / cumsum(e^{e_j-M}) with global row max M
// (softmax shift-invariance). One 128-thread block per batch row.
// ---------------------------------------------------------------------------
__global__ void attn_kernel(const float* __restrict__ h,  // [kB*kT][128]
                            const float* __restrict__ w,  // [128]
                            const float* __restrict__ wb, // [1]
                            float* __restrict__ out) {    // [kB*kT][128]
    const int b   = blockIdx.x;
    const int tid = threadIdx.x;
    __shared__ float e[kT];
    __shared__ float red[128];
    __shared__ float ws[128];
    ws[tid] = w[tid];
    __syncthreads();

    float lmax = -3.4e38f;
#pragma unroll
    for (int p = 0; p < 4; ++p) {
        int t = tid + p * 128;
        const float* hr = h + ((size_t)b * kT + t) * kH;
        float dot = wb[0];
        for (int k = 0; k < 128; ++k) dot += hr[k] * ws[k];
        e[t] = dot;
        lmax = fmaxf(lmax, dot);
    }
    red[tid] = lmax;
    __syncthreads();
    for (int off = 64; off > 0; off >>= 1) {
        if (tid < off) red[tid] = fmaxf(red[tid], red[tid + off]);
        __syncthreads();
    }
    const float M = red[0];

    float S = 0.0f, V = 0.0f;
    for (int t = 0; t < kT; ++t) {
        float wt = __expf(e[t] - M);
        S += wt;
        V += wt * h[((size_t)b * kT + t) * kH + tid];
        out[((size_t)b * kT + t) * kH + tid] = V / S;
    }
}

// ---------------------------------------------------------------------------
static inline int cdiv(long long a, int b) { return (int)((a + b - 1) / b); }

extern "C" void kernel_launch(void* const* d_in, const int* in_sizes, int n_in,
                              void* d_out, int out_size, void* d_ws, size_t ws_size,
                              hipStream_t stream) {
    (void)in_sizes; (void)n_in; (void)out_size; (void)ws_size;

    const float* input_batch = (const float*)d_in[0];
    const float* hgb         = (const float*)d_in[1];
    const float* stat        = (const float*)d_in[2];
    const float* nnan        = (const float*)d_in[3];
    const int*   ttd         = (const int*)d_in[4];
    const float* rte_emb     = (const float*)d_in[5];
    const float* rte_W       = (const float*)d_in[6];
    const float* rte_b       = (const float*)d_in[7];
    const float* sh_Wih = (const float*)d_in[8];
    const float* sh_Whh = (const float*)d_in[9];
    const float* sh_bih = (const float*)d_in[10];
    const float* sh_bhh = (const float*)d_in[11];
    const float* t1_Wih = (const float*)d_in[12];
    const float* t1_Whh = (const float*)d_in[13];
    const float* t1_bih = (const float*)d_in[14];
    const float* t1_bhh = (const float*)d_in[15];
    const float* t2_Wih = (const float*)d_in[16];
    const float* t2_Whh = (const float*)d_in[17];
    const float* t2_bih = (const float*)d_in[18];
    const float* t2_bhh = (const float*)d_in[19];
    const float* t1_w   = (const float*)d_in[20];
    const float* t1_wb  = (const float*)d_in[21];
    const float* t2_w   = (const float*)d_in[22];
    const float* t2_wb  = (const float*)d_in[23];

    float* out = (float*)d_out;
    char* ws = (char*)d_ws;
    float*    tt    = (float*)(ws + OFF_TT);
    _Float16* x16   = (_Float16*)(ws + OFF_X16);
    float*    gi    = (float*)(ws + OFF_GI);
    float*    h_sh  = (float*)(ws + OFF_HSH);
    float*    h_t   = (float*)(ws + OFF_HT);
    _Float16* wih16 = (_Float16*)(ws + OFF_WIH16);
    _Float16* whh16 = (_Float16*)(ws + OFF_WHH16);

    // 1) temporal encoding
    rte_kernel<<<cdiv(kM, 256), 256, 0, stream>>>(ttd, rte_emb, rte_W, rte_b, tt);

    // ---- shared GRU: scans over B (256 steps), 512 independent sequences ----
    pack_w_kernel<<<cdiv((long long)kG * kKshP, 256), 256, 0, stream>>>(
        sh_Wih, wih16, kKsh, kKshP, kG);
    pack_w_kernel<<<cdiv((long long)kG * kH, 256), 256, 0, stream>>>(
        sh_Whh, whh16, kH, kH, kG);
    pack_x_kernel<<<cdiv((long long)kM * kKshP, 256), 256, 0, stream>>>(
        0, kKshP, input_batch, hgb, stat, nnan, tt, h_sh, x16);
    wmma_gemm_kernel<<<dim3(kM / 16, kG / 16), 32, 0, stream>>>(
        x16, wih16, sh_bih, gi, kKshP);
    gru_scan_kernel<<<kT / 16, 256, 0, stream>>>(
        gi, whh16, sh_bhh, h_sh, kB, /*seq_rs=*/1, /*step_rs=*/kT);

    // ---- task GRU 1: scans over T (512 steps), 256 sequences ----
    pack_w_kernel<<<cdiv((long long)kG * kKt1P, 256), 256, 0, stream>>>(
        t1_Wih, wih16, kKt1, kKt1P, kG);
    pack_w_kernel<<<cdiv((long long)kG * kH, 256), 256, 0, stream>>>(
        t1_Whh, whh16, kH, kH, kG);
    pack_x_kernel<<<cdiv((long long)kM * kKt1P, 256), 256, 0, stream>>>(
        1, kKt1P, input_batch, hgb, stat, nnan, tt, h_sh, x16);
    wmma_gemm_kernel<<<dim3(kM / 16, kG / 16), 32, 0, stream>>>(
        x16, wih16, t1_bih, gi, kKt1P);
    gru_scan_kernel<<<kB / 16, 256, 0, stream>>>(
        gi, whh16, t1_bhh, h_t, kT, /*seq_rs=*/kT, /*step_rs=*/1);
    attn_kernel<<<kB, 128, 0, stream>>>(h_t, t1_w, t1_wb, out);

    // ---- task GRU 2 ----
    pack_w_kernel<<<cdiv((long long)kG * kKt2P, 256), 256, 0, stream>>>(
        t2_Wih, wih16, kKt2, kKt2P, kG);
    pack_w_kernel<<<cdiv((long long)kG * kH, 256), 256, 0, stream>>>(
        t2_Whh, whh16, kH, kH, kG);
    pack_x_kernel<<<cdiv((long long)kM * kKt2P, 256), 256, 0, stream>>>(
        2, kKt2P, input_batch, hgb, stat, nnan, tt, h_sh, x16);
    wmma_gemm_kernel<<<dim3(kM / 16, kG / 16), 32, 0, stream>>>(
        x16, wih16, t2_bih, gi, kKt2P);
    gru_scan_kernel<<<kB / 16, 256, 0, stream>>>(
        gi, whh16, t2_bhh, h_t, kT, /*seq_rs=*/kT, /*step_rs=*/1);
    attn_kernel<<<kB, 128, 0, stream>>>(h_t, t2_w, t2_wb, out + (size_t)kM * kH);
}
